// PiNet_16234976379314
// MI455X (gfx1250) — matrix operand order repaired and checked by reading, not compiled, not run on hardware
//
#include <hip/hip_runtime.h>
#include <hip/hip_bf16.h>

typedef __attribute__((ext_vector_type(16))) _Float16 v16h;
typedef __attribute__((ext_vector_type(8)))  float    v8f;

#define NNODES 100000
#define NEDGES 3200000
#define NGRAPH 100
#define NPG    1000
#define D_INF  128
#define D_HF   64
#define D_EF   32
#define N_CLS  10

// ---------------------------------------------------------------- utilities
__global__ void zero_f32(float* __restrict__ p, long n) {
  long i = (long)blockIdx.x * blockDim.x + threadIdx.x;
  long stride = (long)gridDim.x * blockDim.x;
  for (; i < n; i += stride) p[i] = 0.0f;
}

__global__ void deg_kernel(const int* __restrict__ dst, float* __restrict__ deg, int E) {
  int e = blockIdx.x * blockDim.x + threadIdx.x;
  if (e < E) atomicAdd(&deg[dst[e]], 1.0f);
}

__global__ void dinv_kernel(float* __restrict__ dinv, int n) {
  int v = blockIdx.x * blockDim.x + threadIdx.x;
  if (v < n) dinv[v] = rsqrtf(dinv[v] + 2.0f);   // improved=True: +2 self loop
}

// ------------------------------------------------- WMMA 16x16 tile (f16 in, f32 acc)
// A: [M,K] row-major f32 (cvt to f16), W: [K,NOUT] row-major f32, C: [M,NOUT] f32
// A-frag per-lane: two contiguous 8-float runs -> four aligned float4 loads.
__device__ __forceinline__ void wmma_tile(const float* __restrict__ A,
                                          const float* __restrict__ W,
                                          float* __restrict__ C,
                                          int m0, int n0, int K, int NOUT) {
  const int lane = threadIdx.x & 31;
  const int nr = lane & 15;     // M-row for A, N-col for B/C
  const int h  = lane >> 4;
  v8f acc = {};
  const float* arow = A + (size_t)(m0 + nr) * K + (h << 3);
  for (int k0 = 0; k0 < K; k0 += 32) {
    const float* p = arow + k0;
    float4 q0 = *(const float4*)(p);        // K = h*8 + 0..3
    float4 q1 = *(const float4*)(p + 4);    // K = h*8 + 4..7
    float4 q2 = *(const float4*)(p + 16);   // K = 16 + h*8 + 0..3
    float4 q3 = *(const float4*)(p + 20);   // K = 16 + h*8 + 4..7
    v16h a, b;
    a[0]  = (_Float16)q0.x; a[1]  = (_Float16)q0.y;
    a[2]  = (_Float16)q0.z; a[3]  = (_Float16)q0.w;
    a[4]  = (_Float16)q1.x; a[5]  = (_Float16)q1.y;
    a[6]  = (_Float16)q1.z; a[7]  = (_Float16)q1.w;
    a[8]  = (_Float16)q2.x; a[9]  = (_Float16)q2.y;
    a[10] = (_Float16)q2.z; a[11] = (_Float16)q2.w;
    a[12] = (_Float16)q3.x; a[13] = (_Float16)q3.y;
    a[14] = (_Float16)q3.z; a[15] = (_Float16)q3.w;
#pragma unroll
    for (int i = 0; i < 16; ++i)   // B 32x16 layout: K = (i>>3)*16 + h*8 + (i&7), N = lane&15
      b[i] = (_Float16)W[(size_t)(k0 + ((i >> 3) << 4) + (h << 3) + (i & 7)) * NOUT + n0 + nr];
    acc = __builtin_amdgcn_wmma_f32_16x16x32_f16(false, a, false, b,
                                                 (short)0, acc, false, false);
  }
#pragma unroll
  for (int r = 0; r < 8; ++r)      // C/D: M = r + 8*h, N = lane&15
    C[(size_t)(m0 + r + 8 * h) * NOUT + n0 + nr] = acc[r];
}

// Two GEMMs sharing shape: job < NT -> (A0,W0,C0), else (A1,W1,C1)
template <int K, int NOUT>
__global__ void gemm_dual(const float* __restrict__ A0, const float* __restrict__ W0,
                          float* __restrict__ C0,
                          const float* __restrict__ A1, const float* __restrict__ W1,
                          float* __restrict__ C1, int Mtiles) {
  const int NT = NOUT / 16;
  int wave = blockIdx.x * (blockDim.x >> 5) + (threadIdx.x >> 5);
  int mtile = wave % Mtiles;
  int job   = wave / Mtiles;
  if (job >= 2 * NT) return;      // wave-uniform
  const float* A = (job < NT) ? A0 : A1;
  const float* W = (job < NT) ? W0 : W1;
  float*       C = (job < NT) ? C0 : C1;
  wmma_tile(A, W, C, mtile * 16, (job % NT) * 16, K, NOUT);
}

// ------------------------------------------------- edge scatter (both paths fused)
template <int D>
__global__ void scatter_dual(const int* __restrict__ src, const int* __restrict__ dst,
                             const float* __restrict__ dinv,
                             const float* __restrict__ ha, const float* __restrict__ hx,
                             float* __restrict__ aa, float* __restrict__ ax, int E) {
  int e = blockIdx.x * blockDim.x + threadIdx.x;
  if (e >= E) return;
  int s = src[e], d = dst[e];
  float coef = dinv[s] * dinv[d];
  const float4* pa = (const float4*)(ha + (size_t)s * D);
  const float4* px = (const float4*)(hx + (size_t)s * D);
  float* oa = aa + (size_t)d * D;
  float* ox = ax + (size_t)d * D;
#pragma unroll
  for (int c = 0; c < D / 4; ++c) {
    float4 va = pa[c];
    atomicAdd(&oa[c * 4 + 0], va.x * coef);
    atomicAdd(&oa[c * 4 + 1], va.y * coef);
    atomicAdd(&oa[c * 4 + 2], va.z * coef);
    atomicAdd(&oa[c * 4 + 3], va.w * coef);
    float4 vx = px[c];
    atomicAdd(&ox[c * 4 + 0], vx.x * coef);
    atomicAdd(&ox[c * 4 + 1], vx.y * coef);
    atomicAdd(&ox[c * 4 + 2], vx.z * coef);
    atomicAdd(&ox[c * 4 + 3], vx.w * coef);
  }
}

// agg += 2*dinv^2*h + b, optional relu (in-place into agg)
template <int D, bool RELU_A, bool RELU_X>
__global__ void postagg(const float* __restrict__ dinv,
                        const float* __restrict__ ha, const float* __restrict__ hx,
                        const float* __restrict__ ba, const float* __restrict__ bx,
                        float* __restrict__ aa, float* __restrict__ ax, int n) {
  long i = (long)blockIdx.x * blockDim.x + threadIdx.x;
  if (i >= (long)n * D) return;
  int v = (int)(i / D), f = (int)(i % D);
  float self = 2.0f * dinv[v] * dinv[v];
  float va = aa[i] + self * ha[i] + ba[f];
  float vx = ax[i] + self * hx[i] + bx[f];
  aa[i] = RELU_A ? fmaxf(va, 0.0f) : va;
  ax[i] = RELU_X ? fmaxf(vx, 0.0f) : vx;
}

// ------------------------------------------------- per-graph per-feature softmax
__global__ void seg_softmax(float* __restrict__ g2) {
  __shared__ float red[256];
  const int g = blockIdx.x;
  const int tid = threadIdx.x;
  const int f = tid & 31;        // feature 0..31
  const int r0 = tid >> 5;       // row group 0..7
  float* base = g2 + (size_t)g * NPG * D_EF;
  float m = -3.4e38f;
  for (int n = r0; n < NPG; n += 8) m = fmaxf(m, base[n * D_EF + f]);
  red[tid] = m; __syncthreads();
  for (int s = 4; s >= 1; s >>= 1) {
    if (r0 < s) red[tid] = fmaxf(red[tid], red[tid + s * 32]);
    __syncthreads();
  }
  float mf = red[f];
  __syncthreads();
  float ssum = 0.0f;
  for (int n = r0; n < NPG; n += 8) ssum += expf(base[n * D_EF + f] - mf);
  red[tid] = ssum; __syncthreads();
  for (int s = 4; s >= 1; s >>= 1) {
    if (r0 < s) red[tid] += red[tid + s * 32];
    __syncthreads();
  }
  float inv = 1.0f / red[f];
  __syncthreads();
  for (int n = r0; n < NPG; n += 8)
    base[n * D_EF + f] = expf(base[n * D_EF + f] - mf) * inv;
}

// ------------------------------------------------- per-graph A^T X via WMMA
// prods[g, e, f] = sum_n a2[g,n,e] * x2[g,n,f] ; one wave per 16x16 quadrant.
// Tail guard uses address clamp + value select (no EXEC divergence, bulk loads).
__global__ void bmm_wmma(const float* __restrict__ a2, const float* __restrict__ x2,
                         float* __restrict__ prods) {
  const int g = blockIdx.x;
  const int w = threadIdx.x >> 5;       // 0..3 quadrant
  const int lane = threadIdx.x & 31;
  const int nr = lane & 15;
  const int h = lane >> 4;
  const int e0 = (w & 1) * 16;
  const int f0 = (w >> 1) * 16;
  const float* A = a2 + (size_t)g * NPG * D_EF;
  const float* X = x2 + (size_t)g * NPG * D_EF;
  v8f acc = {};
  for (int n0 = 0; n0 < NPG; n0 += 32) {   // 32 steps, K padded 1000 -> 1024
    v16h a, b;
#pragma unroll
    for (int i = 0; i < 16; ++i) {
      int k = n0 + ((i >> 3) << 4) + (h << 3) + (i & 7);
      int kc = (k < NPG) ? k : (NPG - 1);            // always-valid address
      float v = A[(size_t)kc * D_EF + e0 + nr];      // unconditional load
      a[i] = (_Float16)((k < NPG) ? v : 0.0f);       // v_cndmask on value
    }
#pragma unroll
    for (int i = 0; i < 16; ++i) {
      int k = n0 + ((i >> 3) << 4) + (h << 3) + (i & 7);
      int kc = (k < NPG) ? k : (NPG - 1);
      float v = X[(size_t)kc * D_EF + f0 + nr];
      b[i] = (_Float16)((k < NPG) ? v : 0.0f);
    }
    acc = __builtin_amdgcn_wmma_f32_16x16x32_f16(false, a, false, b,
                                                 (short)0, acc, false, false);
  }
#pragma unroll
  for (int r = 0; r < 8; ++r)
    prods[(size_t)g * (D_EF * D_EF) + (e0 + r + 8 * h) * D_EF + f0 + nr] = acc[r];
}

// ------------------------------------------------- 1024->10 linear + softmax
__global__ void final_linear_softmax(const float* __restrict__ prods,
                                     const float* __restrict__ Wl,
                                     const float* __restrict__ bl,
                                     float* __restrict__ out) {
  const int g = blockIdx.x;
  const int lane = threadIdx.x;   // blockDim = 32 (one wave)
  const float* p = prods + (size_t)g * (D_EF * D_EF);
  float acc[N_CLS];
#pragma unroll
  for (int c = 0; c < N_CLS; ++c) acc[c] = 0.0f;
  for (int k = lane; k < D_EF * D_EF; k += 32) {
    float pv = p[k];
    const float* wr = Wl + (size_t)k * N_CLS;
#pragma unroll
    for (int c = 0; c < N_CLS; ++c) acc[c] += pv * wr[c];
  }
#pragma unroll
  for (int c = 0; c < N_CLS; ++c)
    for (int off = 16; off > 0; off >>= 1)
      acc[c] += __shfl_down(acc[c], off, 32);
  if (lane == 0) {
    float logit[N_CLS], m = -3.4e38f;
    for (int c = 0; c < N_CLS; ++c) { logit[c] = acc[c] + bl[c]; m = fmaxf(m, logit[c]); }
    float s = 0.0f;
    for (int c = 0; c < N_CLS; ++c) { logit[c] = expf(logit[c] - m); s += logit[c]; }
    float inv = 1.0f / s;
    for (int c = 0; c < N_CLS; ++c) out[(size_t)g * N_CLS + c] = logit[c] * inv;
  }
}

// ---------------------------------------------------------------- launcher
extern "C" void kernel_launch(void* const* d_in, const int* in_sizes, int n_in,
                              void* d_out, int out_size, void* d_ws, size_t ws_size,
                              hipStream_t stream) {
  (void)in_sizes; (void)n_in; (void)out_size; (void)ws_size;
  const float* x     = (const float*)d_in[0];
  const int*   eidx  = (const int*)d_in[1];
  // d_in[2] = batch (unused: uniform graphs -> graph = node / NPG)
  const float* W_a1  = (const float*)d_in[3];
  const float* b_a1  = (const float*)d_in[4];
  const float* W_a2  = (const float*)d_in[5];
  const float* b_a2  = (const float*)d_in[6];
  const float* W_x1  = (const float*)d_in[7];
  const float* b_x1  = (const float*)d_in[8];
  const float* W_x2  = (const float*)d_in[9];
  const float* b_x2  = (const float*)d_in[10];
  const float* W_lin = (const float*)d_in[11];
  const float* b_lin = (const float*)d_in[12];
  float* out = (float*)d_out;

  const int* src = eidx;             // edge_index[0,:]
  const int* dst = eidx + NEDGES;    // edge_index[1,:]

  float* ws = (float*)d_ws;
  size_t off = 0;
  float* dinv  = ws + off; off += NNODES;
  float* h_a   = ws + off; off += (size_t)NNODES * D_HF;
  float* h_x   = ws + off; off += (size_t)NNODES * D_HF;
  float* agg_a = ws + off; off += (size_t)NNODES * D_HF;  // becomes a1
  float* agg_x = ws + off; off += (size_t)NNODES * D_HF;  // becomes x1
  float* h_a2  = ws + off; off += (size_t)NNODES * D_EF;
  float* h_x2  = ws + off; off += (size_t)NNODES * D_EF;
  float* agg2a = ws + off; off += (size_t)NNODES * D_EF;  // becomes a2 (softmax in place)
  float* agg2x = ws + off; off += (size_t)NNODES * D_EF;  // becomes x2
  float* prods = ws + off; off += (size_t)NGRAPH * D_EF * D_EF;

  const int Mtiles = NNODES / 16;    // 6250

  // degree / dinv (shared by all four GCN layers)
  zero_f32<<<1024, 256, 0, stream>>>(dinv, NNODES);
  deg_kernel<<<(NEDGES + 255) / 256, 256, 0, stream>>>(dst, dinv, NEDGES);
  dinv_kernel<<<(NNODES + 255) / 256, 256, 0, stream>>>(dinv, NNODES);

  // layer 1 dense: h_a = x@W_a1, h_x = x@W_x1   (waves = Mtiles*8, 4 waves/block)
  gemm_dual<D_INF, D_HF><<<Mtiles * 8 / 4, 128, 0, stream>>>(x, W_a1, h_a, x, W_x1, h_x, Mtiles);

  // layer 1 aggregate
  zero_f32<<<4096, 256, 0, stream>>>(agg_a, (long)NNODES * D_HF);
  zero_f32<<<4096, 256, 0, stream>>>(agg_x, (long)NNODES * D_HF);
  scatter_dual<D_HF><<<(NEDGES + 255) / 256, 256, 0, stream>>>(src, dst, dinv, h_a, h_x, agg_a, agg_x, NEDGES);
  postagg<D_HF, true, true><<<(int)(((long)NNODES * D_HF + 255) / 256), 256, 0, stream>>>(
      dinv, h_a, h_x, b_a1, b_x1, agg_a, agg_x, NNODES);

  // layer 2 dense: h_a2 = a1@W_a2, h_x2 = x1@W_x2
  gemm_dual<D_HF, D_EF><<<Mtiles * 4 / 4, 128, 0, stream>>>(agg_a, W_a2, h_a2, agg_x, W_x2, h_x2, Mtiles);

  // layer 2 aggregate (a-path: no relu; x-path: relu)
  zero_f32<<<4096, 256, 0, stream>>>(agg2a, (long)NNODES * D_EF);
  zero_f32<<<4096, 256, 0, stream>>>(agg2x, (long)NNODES * D_EF);
  scatter_dual<D_EF><<<(NEDGES + 255) / 256, 256, 0, stream>>>(src, dst, dinv, h_a2, h_x2, agg2a, agg2x, NEDGES);
  postagg<D_EF, false, true><<<(int)(((long)NNODES * D_EF + 255) / 256), 256, 0, stream>>>(
      dinv, h_a2, h_x2, b_a2, b_x2, agg2a, agg2x, NNODES);

  // segment softmax over each graph's 1000 nodes, per feature (in place)
  seg_softmax<<<NGRAPH, 256, 0, stream>>>(agg2a);

  // prods[b] = a2_b^T @ x2_b  (WMMA, 4 waves per graph)
  bmm_wmma<<<NGRAPH, 128, 0, stream>>>(agg2a, agg2x, prods);

  // classifier + softmax
  final_linear_softmax<<<NGRAPH, 32, 0, stream>>>(prods, W_lin, b_lin, out);
}
